// GCNNet_46505905881182
// MI455X (gfx1250) — matrix operand order repaired and checked by reading, not compiled, not run on hardware
//
#include <hip/hip_runtime.h>
#include <hip/hip_bf16.h>

typedef __attribute__((ext_vector_type(16))) __bf16 v16bf;
typedef __attribute__((ext_vector_type(8)))  __bf16 v8bf;
typedef __attribute__((ext_vector_type(8)))  float  v8f;

#define RELU_IN  1
#define RELU_OUT 2

#define CAT16(lo, hi) __builtin_shufflevector(lo, hi, 0,1,2,3,4,5,6,7,8,9,10,11,12,13,14,15)

__device__ __forceinline__ v8f wmma_bf16(v16bf a, v16bf b, v8f c) {
    return __builtin_amdgcn_wmma_f32_16x16x32_bf16(false, a, false, b, (short)0, c,
                                                   false, false);
}

// ---------------------------------------------------------------------------
// Generic GEMM: C[M,ldc] = act(A[M,K] @ B[K,N] + bias), bf16 WMMA, f32 accum.
// Block tile 64x32, K-step 32, 256 threads = 8 waves, one 16x16 tile per wave.
// Fragment loads are 2x ds_load_b128 each (A rows contiguous; B transposed).
// ---------------------------------------------------------------------------
__global__ __launch_bounds__(256)
void wmma_gemm_kernel(const float* __restrict__ A, const float* __restrict__ B,
                      const float* __restrict__ bias, float* __restrict__ C,
                      int M, int N, int K, int ldc, int flags)
{
    __shared__ alignas(16) __bf16 As[64][40];     // [m][k], 80B row stride
    __shared__ alignas(16) __bf16 BsT[32][40];    // [n][k] (transposed)

    const bool reluIn  = (flags & RELU_IN)  != 0;
    const bool reluOut = (flags & RELU_OUT) != 0;

    const int tid  = threadIdx.x;
    const int lane = tid & 31, wid = tid >> 5;
    const int half = lane >> 4, mr = lane & 15;
    const int rowW = (wid & 3) * 16;              // 4 wave-tiles along M
    const int colW = (wid >> 2) * 16;             // 2 wave-tiles along N
    const long long rowBase = (long long)blockIdx.x * 64;
    const int  colBase = blockIdx.y * 32;

    v8f acc = {0.f, 0.f, 0.f, 0.f, 0.f, 0.f, 0.f, 0.f};

    const __bf16* aRow = &As[rowW + mr][0];
    const __bf16* bRow = &BsT[colW + mr][0];

    for (int kb = 0; kb < K; kb += 32) {
        for (int i = tid; i < 64 * 32; i += 256) {
            int r = i >> 5, c = i & 31;
            long long gr = rowBase + r; int gc = kb + c;
            float v = 0.f;
            if (gr < M && gc < K) v = A[gr * K + gc];
            if (reluIn) v = fmaxf(v, 0.f);
            As[r][c] = (__bf16)v;
        }
        for (int i = tid; i < 32 * 32; i += 256) {
            int r = i >> 5, c = i & 31;          // r = k, c = n
            int gr = kb + r, gc = colBase + c;
            float v = 0.f;
            if (gr < K && gc < N) v = B[(long long)gr * N + gc];
            BsT[c][r] = (__bf16)v;
        }
        // prefetch next A K-tile while this one is consumed
        if (kb + 32 < K) {
            long long gr = rowBase + (tid >> 2);
            int gc = kb + 32 + (tid & 3) * 8;
            if (gr < M && gc < K) __builtin_prefetch(&A[gr * K + gc], 0, 0);
        }
        __syncthreads();
        v8bf alo = *(const v8bf*)(aRow + 8 * half);
        v8bf ahi = *(const v8bf*)(aRow + 16 + 8 * half);
        v8bf blo = *(const v8bf*)(bRow + 8 * half);
        v8bf bhi = *(const v8bf*)(bRow + 16 + 8 * half);
        acc = wmma_bf16(CAT16(alo, ahi), CAT16(blo, bhi), acc);
        __syncthreads();
    }
#pragma unroll
    for (int r = 0; r < 8; ++r) {
        long long gm = rowBase + rowW + 8 * half + r;
        int gn = colBase + colW + mr;
        if (gm < M && gn < N) {
            float v = acc[r];
            if (bias) v += bias[gn];
            if (reluOut) v = fmaxf(v, 0.f);
            C[gm * ldc + gn] = v;
        }
    }
}

// ---------------------------------------------------------------------------
// Degree / normalization
// ---------------------------------------------------------------------------
__global__ void deg_init_kernel(float* deg, int N) {
    int i = blockIdx.x * blockDim.x + threadIdx.x;
    if (i < N) deg[i] = 1.0f;                     // self loop
}
__global__ void deg_scatter_kernel(const int* __restrict__ dst, float* deg, int E) {
    int e = blockIdx.x * blockDim.x + threadIdx.x;
    if (e < E)
        __hip_atomic_fetch_add(&deg[dst[e]], 1.0f, __ATOMIC_RELAXED,
                               __HIP_MEMORY_SCOPE_AGENT);
}
__global__ void deg_rsqrt_kernel(float* deg, int N) {
    int i = blockIdx.x * blockDim.x + threadIdx.x;
    if (i < N) deg[i] = rsqrtf(deg[i]);
}

// ---------------------------------------------------------------------------
// GCN aggregation: out[i] = XW[i]*dinv[i]^2 + b  then  out[dst] += XW[src]*norm
// ---------------------------------------------------------------------------
__global__ void gcn_init_kernel(const float* __restrict__ XW, const float* __restrict__ dinv,
                                const float* __restrict__ bias, float* __restrict__ out, int C)
{
    long long i = blockIdx.x;
    float d = dinv[i], d2 = d * d;
    for (int f = threadIdx.x; f < C; f += blockDim.x)
        out[i * C + f] = XW[i * C + f] * d2 + bias[f];
}
__global__ void gcn_edge_kernel(const float* __restrict__ XW, const int* __restrict__ src,
                                const int* __restrict__ dst, const float* __restrict__ dinv,
                                float* __restrict__ out, int C)
{
    int e = blockIdx.x;
    int s = src[e], d = dst[e];
    float nrm = dinv[s] * dinv[d];
    const float* xs = XW + (long long)s * C;
    float* od = out + (long long)d * C;
    for (int f = threadIdx.x; f < C; f += blockDim.x)
        __hip_atomic_fetch_add(&od[f], xs[f] * nrm, __ATOMIC_RELAXED,
                               __HIP_MEMORY_SCOPE_AGENT);
}

// max-pool over 40 contiguous nodes per graph, fused ReLU (max with 0)
__global__ void pool_kernel(const float* __restrict__ h, float* __restrict__ g)
{
    int gi = blockIdx.x;
    for (int f = threadIdx.x; f < 312; f += blockDim.x) {
        float m = 0.f;
        const float* p = h + (long long)gi * 40 * 312 + f;
        for (int n = 0; n < 40; ++n) m = fmaxf(m, p[n * 312]);
        g[gi * 312 + f] = m;
    }
}

// ---------------------------------------------------------------------------
// Conv weight rearrange (transposed): Wc[o][c][k] -> WrT[o*6016 + (c*8+k)],
// bf16, K padded to 6016 with zeros. grid = (ceil(6016/256), 32)
// ---------------------------------------------------------------------------
__global__ void rearrange_wc_kernel(const float* __restrict__ Wc, __bf16* __restrict__ WrT)
{
    int o  = blockIdx.y;
    int ck = blockIdx.x * blockDim.x + threadIdx.x;
    if (ck < 6016)
        WrT[o * 6016 + ck] = (ck < 6000) ? (__bf16)Wc[o * 6000 + ck] : (__bf16)0.f;
}

// ---------------------------------------------------------------------------
// Protein branch 2: out[b][o*121+l] = sum_{c,k} emb[t2[b,c]][l+k]*Wc2[o,c,k] + bc2[o]
// Per-block im2col WMMA GEMM: M=121(->128), N=32, K=6000(->6016). emb+t2 in LDS.
// Each 8-element K-run of the A fragment is one conv channel (contiguous in LDS);
// rows l>=121 compute garbage from the pad region and are masked at the store.
// ---------------------------------------------------------------------------
__global__ __launch_bounds__(256)
void conv_emb_kernel(const float* __restrict__ emb, const int* __restrict__ t2,
                     const __bf16* __restrict__ WrT, const float* __restrict__ bias,
                     float* __restrict__ out)
{
    __shared__ alignas(16) __bf16 embs[26 * 128 + 16];   // +16 pad for masked rows
    __shared__ int t2s[750];
    int b = blockIdx.x, tid = threadIdx.x;
    for (int i = tid; i < 26 * 128 + 16; i += 256) embs[i] = (i < 26 * 128) ? (__bf16)emb[i] : (__bf16)0.f;
    for (int i = tid; i < 750; i += 256) t2s[i] = t2[b * 750 + i];
    __syncthreads();

    const int lane = tid & 31, wid = tid >> 5;
    const int half = lane >> 4, mr = lane & 15;
    const int l = wid * 16 + mr;                  // output position for this lane

    const __bf16* w0 = WrT + (size_t)mr * 6016;          // o = mr
    const __bf16* w1 = WrT + (size_t)(16 + mr) * 6016;   // o = 16+mr

    v8f acc0 = {0.f,0.f,0.f,0.f,0.f,0.f,0.f,0.f};
    v8f acc1 = {0.f,0.f,0.f,0.f,0.f,0.f,0.f,0.f};

    for (int kb = 0; kb < 6016; kb += 32) {
        int cb = kb >> 3;
        int c0 = cb + half, c1 = cb + 2 + half;
        v16bf a;
        if (c0 < 750) {
            int base = t2s[c0] * 128 + l;
#pragma unroll
            for (int t = 0; t < 8; ++t) a[t] = embs[base + t];
        } else {
#pragma unroll
            for (int t = 0; t < 8; ++t) a[t] = (__bf16)0.f;
        }
        if (c1 < 750) {
            int base = t2s[c1] * 128 + l;
#pragma unroll
            for (int t = 0; t < 8; ++t) a[8 + t] = embs[base + t];
        } else {
#pragma unroll
            for (int t = 0; t < 8; ++t) a[8 + t] = (__bf16)0.f;
        }
        int ko = kb + 8 * half;
        v16bf b0 = CAT16(*(const v8bf*)(w0 + ko), *(const v8bf*)(w0 + ko + 16));
        v16bf b1 = CAT16(*(const v8bf*)(w1 + ko), *(const v8bf*)(w1 + ko + 16));
        acc0 = wmma_bf16(a, b0, acc0);
        acc1 = wmma_bf16(a, b1, acc1);
    }
#pragma unroll
    for (int r = 0; r < 8; ++r) {
        int ll = wid * 16 + 8 * half + r;
        if (ll < 121) {
            out[(long long)b * 3872 + mr * 121 + ll]        = acc0[r] + bias[mr];
            out[(long long)b * 3872 + (16 + mr) * 121 + ll] = acc1[r] + bias[16 + mr];
        }
    }
}

// ---------------------------------------------------------------------------
// Protein branch 1: out[b][o*13+l] = sum_{c,k} t1[b,c,l+k]*Wc1[o,c,k] + bc1[o]
// M=13(->16), N=32, K=6016. Protein row staged as bf16 in LDS (30 KB).
// Rows l>=13 read pad/garbage and are masked at the store.
// ---------------------------------------------------------------------------
__global__ __launch_bounds__(256)
void conv_t1_kernel(const float* __restrict__ t1, const __bf16* __restrict__ WrT,
                    const float* __restrict__ bias, float* __restrict__ out)
{
    __shared__ alignas(16) __bf16 t1s[750 * 20 + 16];
    int b = blockIdx.x, tid = threadIdx.x;
    for (int i = tid; i < 15000 + 16; i += 256)
        t1s[i] = (i < 15000) ? (__bf16)t1[(long long)b * 15000 + i] : (__bf16)0.f;
    __syncthreads();

    const int lane = tid & 31, wid = tid >> 5;
    if (wid < 2) {                                // 2 waves = 2 N-tiles of 16
        const int half = lane >> 4, mr = lane & 15;
        const int ob = wid * 16;
        const __bf16* wb = WrT + (size_t)(ob + mr) * 6016;
        v8f acc = {0.f,0.f,0.f,0.f,0.f,0.f,0.f,0.f};

        for (int kb = 0; kb < 6016; kb += 32) {
            int cb = kb >> 3;
            int c0 = cb + half, c1 = cb + 2 + half;
            v16bf a;
            if (c0 < 750) {
                int base = c0 * 20 + mr;
#pragma unroll
                for (int t = 0; t < 8; ++t) a[t] = t1s[base + t];
            } else {
#pragma unroll
                for (int t = 0; t < 8; ++t) a[t] = (__bf16)0.f;
            }
            if (c1 < 750) {
                int base = c1 * 20 + mr;
#pragma unroll
                for (int t = 0; t < 8; ++t) a[8 + t] = t1s[base + t];
            } else {
#pragma unroll
                for (int t = 0; t < 8; ++t) a[8 + t] = (__bf16)0.f;
            }
            int ko = kb + 8 * half;
            v16bf bb = CAT16(*(const v8bf*)(wb + ko), *(const v8bf*)(wb + ko + 16));
            acc = wmma_bf16(a, bb, acc);
        }
#pragma unroll
        for (int r = 0; r < 8; ++r) {
            int ll = 8 * half + r;
            if (ll < 13)
                out[(long long)b * 416 + (ob + mr) * 13 + ll] = acc[r] + bias[ob + mr];
        }
    }
}

// final head: out[b] = f2[b,:] . Wo + bo   (one wave32 per row)
__global__ void head_kernel(const float* __restrict__ f2, const float* __restrict__ Wo,
                            const float* __restrict__ bo, float* __restrict__ out)
{
    int b = blockIdx.x, lane = threadIdx.x;
    float s = 0.f;
    for (int f = lane; f < 512; f += 32) s += f2[(long long)b * 512 + f] * Wo[f];
#pragma unroll
    for (int o = 16; o > 0; o >>= 1) s += __shfl_down(s, o, 32);
    if (lane == 0) out[b] = s + bo[0];
}

// ---------------------------------------------------------------------------
extern "C" void kernel_launch(void* const* d_in, const int* in_sizes, int n_in,
                              void* d_out, int out_size, void* d_ws, size_t ws_size,
                              hipStream_t stream)
{
    (void)in_sizes; (void)n_in; (void)out_size; (void)ws_size;

    const float* x    = (const float*)d_in[0];
    const int*   ei   = (const int*)  d_in[1];
    const float* t1   = (const float*)d_in[3];
    const int*   t2   = (const int*)  d_in[4];
    const float* W1   = (const float*)d_in[5];  const float* b1  = (const float*)d_in[6];
    const float* W2   = (const float*)d_in[7];  const float* b2  = (const float*)d_in[8];
    const float* W3   = (const float*)d_in[9];  const float* b3  = (const float*)d_in[10];
    const float* Wg1  = (const float*)d_in[11]; const float* bg1 = (const float*)d_in[12];
    const float* Wg2  = (const float*)d_in[13]; const float* bg2 = (const float*)d_in[14];
    const float* emb  = (const float*)d_in[15];
    const float* Wc2  = (const float*)d_in[16]; const float* bc2 = (const float*)d_in[17];
    const float* Wxt2 = (const float*)d_in[18]; const float* bxt2= (const float*)d_in[19];
    const float* Wc1  = (const float*)d_in[20]; const float* bc1 = (const float*)d_in[21];
    const float* Wxt1 = (const float*)d_in[22]; const float* bxt1= (const float*)d_in[23];
    const float* Wf1  = (const float*)d_in[24]; const float* bf1 = (const float*)d_in[25];
    const float* Wf2  = (const float*)d_in[26]; const float* bf2 = (const float*)d_in[27];
    const float* Wo   = (const float*)d_in[28]; const float* bo  = (const float*)d_in[29];
    float* out = (float*)d_out;

    const int N = 40960, E = 163840;
    const int* srcIdx = ei;
    const int* dstIdx = ei + E;

    // -------- workspace carving (~154 MB) --------
    char* ws = (char*)d_ws;
    size_t off = 0;
    auto carve = [&](size_t bytes) -> char* {
        off = (off + 255) & ~(size_t)255;
        char* p = ws + off; off += bytes; return p;
    };
    float* dinv = (float*)carve((size_t)N * 4);
    char*  R0   = carve((size_t)N * 312 * 4);   // XW, later conv buffers
    char*  R1   = carve((size_t)N * 312 * 4);   // aggA, later f1/f2
    char*  R2   = carve((size_t)N * 312 * 4);   // aggB, later gpool/gh1/xc

    float*  XW       = (float*)R0;
    float*  aggA     = (float*)R1;
    float*  aggB     = (float*)R2;
    // phase-2 aliases (regions dead by the time these are used)
    float*  gpool    = (float*)(R2);
    float*  gh1      = (float*)(R2 + (2u << 20));
    float*  xc       = (float*)(R2 + (8u << 20));
    float*  conv2out = (float*)R0;
    __bf16* W2rT     = (__bf16*)(R0 + (16u << 20));
    __bf16* W1rT     = (__bf16*)(R0 + (17u << 20));
    float*  conv1out = (float*)(R0 + (18u << 20));
    float*  f1       = (float*)R1;
    float*  f2       = (float*)(R1 + (8u << 20));

    auto gemm = [&](const float* A, const float* Bm, const float* bias, float* C,
                    int M, int Nn, int K, int ldc, int flags) {
        dim3 grid((M + 63) / 64, (Nn + 31) / 32);
        wmma_gemm_kernel<<<grid, 256, 0, stream>>>(A, Bm, bias, C, M, Nn, K, ldc, flags);
    };

    // degrees -> dinv
    deg_init_kernel<<<(N + 255) / 256, 256, 0, stream>>>(dinv, N);
    deg_scatter_kernel<<<(E + 255) / 256, 256, 0, stream>>>(dstIdx, dinv, E);
    deg_rsqrt_kernel<<<(N + 255) / 256, 256, 0, stream>>>(dinv, N);

    // GCN layer 1 (78 -> 78)
    gemm(x, W1, nullptr, XW, N, 78, 78, 78, 0);
    gcn_init_kernel<<<N, 128, 0, stream>>>(XW, dinv, b1, aggA, 78);
    gcn_edge_kernel<<<E, 128, 0, stream>>>(XW, srcIdx, dstIdx, dinv, aggA, 78);
    // GCN layer 2 (78 -> 156), ReLU fused into A-load
    gemm(aggA, W2, nullptr, XW, N, 156, 78, 156, RELU_IN);
    gcn_init_kernel<<<N, 128, 0, stream>>>(XW, dinv, b2, aggB, 156);
    gcn_edge_kernel<<<E, 128, 0, stream>>>(XW, srcIdx, dstIdx, dinv, aggB, 156);
    // GCN layer 3 (156 -> 312)
    gemm(aggB, W3, nullptr, XW, N, 312, 156, 312, RELU_IN);
    gcn_init_kernel<<<N, 128, 0, stream>>>(XW, dinv, b3, aggA, 312);
    gcn_edge_kernel<<<E, 128, 0, stream>>>(XW, srcIdx, dstIdx, dinv, aggA, 312);

    // pool (ReLU fused) and graph head; Wg2 output lands in xc cols [0,128)
    pool_kernel<<<1024, 128, 0, stream>>>(aggA, gpool);
    gemm(gpool, Wg1, bg1, gh1, 1024, 1024, 312, 1024, RELU_OUT);
    gemm(gh1, Wg2, bg2, xc, 1024, 128, 1024, 384, 0);

    // protein branch 2 -> xc cols [256,384)
    {
        dim3 rg((6016 + 255) / 256, 32);
        rearrange_wc_kernel<<<rg, 256, 0, stream>>>(Wc2, W2rT);
    }
    conv_emb_kernel<<<1024, 256, 0, stream>>>(emb, t2, W2rT, bc2, conv2out);
    gemm(conv2out, Wxt2, bxt2, xc + 256, 1024, 128, 3872, 384, 0);

    // protein branch 1 -> xc cols [128,256)
    {
        dim3 rg((6016 + 255) / 256, 32);
        rearrange_wc_kernel<<<rg, 256, 0, stream>>>(Wc1, W1rT);
    }
    conv_t1_kernel<<<1024, 256, 0, stream>>>(t1, W1rT, bc1, conv1out);
    gemm(conv1out, Wxt1, bxt1, xc + 128, 1024, 128, 416, 384, 0);

    // fusion head
    gemm(xc, Wf1, bf1, f1, 1024, 1024, 384, 1024, RELU_OUT);
    gemm(f1, Wf2, bf2, f2, 1024, 512, 1024, 512, RELU_OUT);
    head_kernel<<<1024, 32, 0, stream>>>(f2, Wo, bo, out);
}
